// SimpleBiGRU_37555194036853
// MI455X (gfx1250) — matrix operand order repaired and checked by reading, not compiled, not run on hardware
//
#include <hip/hip_runtime.h>
#include <hip/hip_bf16.h>
#include <math.h>
#include <stdint.h>

// ---------------------------------------------------------------------------
// Bidirectional GRU for MI455X (gfx1250): persistent batch-tile workgroups,
// bf16 WMMA (v_wmma_f32_16x16x32_bf16) with fp32 accumulation, weights
// transposed/converted to bf16 once into d_ws and streamed from L2 each step.
// A laundered 32-bit offset (not a laundered pointer!) makes the weight
// addresses loop-variant -> no 312-VGPR hoist/spill, while keeping the base
// a typed kernarg pointer -> GLOBAL (not FLAT) loads, so weight traffic uses
// LOADcnt only and stays decoupled from the LDS/DScnt path.
// ---------------------------------------------------------------------------

typedef __bf16 bf16_t;
typedef __attribute__((ext_vector_type(16))) __bf16 v16bf;
typedef __attribute__((ext_vector_type(8)))  __bf16 v8bf;
typedef __attribute__((ext_vector_type(8)))  float  v8f;

#define TT 1000     // sequence length
#define BB 256      // batch
#define NN 128      // spike features
#define EE 4        // external features
#define DD 132      // NN + EE
#define DP 160      // DD padded to multiple of 32 (K for input GEMM)
#define HH 256      // hidden
#define G3 768      // 3*HH
#define BT 16       // batch tile per workgroup (one WMMA M-tile)
#define NTHREADS 512
#define NWAVE 16

union ABfrag { v16bf v; v8bf h[2]; };

// ---------------------------------------------------------------------------
// Prep: convert W to bf16 and transpose to [N][K] row-major so each WMMA
// B-fragment lane reads two runs of 8 contiguous bf16, 16 apart in K (two b128 loads).
// ws layout (bf16 elems): WiT_f[G3][DP] | WhT_f[G3][HH] | WiT_b | WhT_b
// ---------------------------------------------------------------------------
__global__ void gru_prep_weights(const float* __restrict__ Wi_f,
                                 const float* __restrict__ Wh_f,
                                 const float* __restrict__ Wi_b,
                                 const float* __restrict__ Wh_b,
                                 bf16_t* __restrict__ ws) {
    const int szWi = G3 * DP;
    const int szWh = G3 * HH;
    const int total = 2 * (szWi + szWh);
    for (int i = blockIdx.x * blockDim.x + threadIdx.x; i < total;
         i += gridDim.x * blockDim.x) {
        int j = i;
        int dir = 0;
        if (j >= szWi + szWh) { dir = 1; j -= szWi + szWh; }
        const float* Wi = dir ? Wi_b : Wi_f;
        const float* Wh = dir ? Wh_b : Wh_f;
        bf16_t v;
        if (j < szWi) {
            int n = j / DP, k = j - n * DP;
            v = (k < DD) ? (bf16_t)Wi[k * G3 + n] : (bf16_t)0.0f;
        } else {
            int jj = j - szWi;
            int n = jj / HH, k = jj - n * HH;
            v = (bf16_t)Wh[k * G3 + n];
        }
        ws[i] = v;
    }
}

// ---------------------------------------------------------------------------
// Main recurrent kernel. grid = 32 blocks: blockIdx.x = {batch_tile, dir}.
// 512 threads = 16 waves; wave w owns hidden columns [16w, 16w+16) in every
// gate block -> accumulators: r (gi+gh fused), z (fused), gi_n, gh_n.
// ---------------------------------------------------------------------------
__global__ __launch_bounds__(NTHREADS)
void gru_bidir_kernel(const float* __restrict__ spike,
                      const float* __restrict__ extin,
                      const int*   __restrict__ lens,
                      const float* __restrict__ bi_f,
                      const float* __restrict__ bhn_f,
                      const float* __restrict__ bi_b,
                      const float* __restrict__ bhn_b,
                      const bf16_t* __restrict__ ws,
                      float* __restrict__ out) {
    __shared__ __align__(16) bf16_t x_lds[BT * DP];   // x_t tile, bf16, K-padded
    __shared__ __align__(16) bf16_t h_lds[BT * HH];   // h (matmul operand, bf16)
    __shared__ __align__(16) float  h32  [BT * HH];   // h master copy (fp32)
    __shared__ int L_lds[BT];

    const int tid  = threadIdx.x;
    const int lane = tid & 31;
    const int w    = tid >> 5;          // wave id 0..15
    const int dir  = blockIdx.x & 1;    // 0 = fwd, 1 = bwd
    const int b0   = (blockIdx.x >> 1) * BT;

    const bf16_t* WiT0 = ws + (dir ? (G3 * DP + G3 * HH) : 0);
    const bf16_t* WhT0 = WiT0 + G3 * DP;
    const float*  bi  = dir ? bi_b  : bi_f;
    const float*  bhn = dir ? bhn_b : bhn_f;

    const int nloc = lane & 15;         // column within 16-wide N tile
    const int mhi  = lane >> 4;         // lane group (selects K half / M half)
    const int hcol = w * 16 + nloc;     // hidden index 0..255 owned by this lane
    const float bir  = bi[hcol];
    const float biz  = bi[HH + hcol];
    const float bin  = bi[2 * HH + hcol];
    const float bhnv = bhn[hcol];

    // --- init: trial lengths + h0 = 0 ---
    for (int i = tid; i < BT; i += NTHREADS) L_lds[i] = lens[b0 + i];
    for (int i = tid; i < BT * HH; i += NTHREADS) {
        h_lds[i] = (bf16_t)0.0f;
        h32[i]   = 0.0f;
    }
    __syncthreads();

    // cooperative x_t tile load (flip folded in for bwd; pad K to DP w/ zeros)
    auto load_x = [&](int t) {
        for (int i = tid; i < BT * DP; i += NTHREADS) {
            int r = i / DP, c = i - r * DP;
            int L = L_lds[r];
            int ts = dir ? ((L + TT - 1 - t) % TT) : t;
            float v = 0.0f;
            if (c < NN)      v = spike[((size_t)(b0 + r) * TT + ts) * NN + c];
            else if (c < DD) v = extin[((size_t)(b0 + r) * TT + ts) * EE + (c - NN)];
            x_lds[i] = (bf16_t)v;
        }
    };
    load_x(0);

    // A fragment (16x32 bf16): row = lane%16; two 8-elem runs at
    // K = kk*32 + {0|8} and +16 per ISA layout.
    auto loadA = [&](const bf16_t* base, int stride, int kk) -> v16bf {
        const bf16_t* p = base + (lane & 15) * stride + kk * 32 + (mhi << 3);
        ABfrag u;
        u.h[0] = *(const v8bf*)p;
        u.h[1] = *(const v8bf*)(p + 16);
        return u.v;
    };
    // B fragment (32x16 bf16): col = lane%16; two 8-elem runs at
    // K = kk*32 + {0|8} and +16 (matching A) from [N][K]-transposed weights.
    auto loadB = [&](const bf16_t* base, int stride, int ncol, int kk) -> v16bf {
        const bf16_t* p = base + (size_t)ncol * stride + kk * 32 + (mhi << 3);
        ABfrag u;
        u.h[0] = *(const v8bf*)p;
        u.h[1] = *(const v8bf*)(p + 16);
        return u.v;
    };

    // Laundered zero offset: loop-variant to the compiler, so the 39 weight
    // fragments (312 VGPRs) cannot be hoisted out of the t-loop and spilled.
    // The pointer itself stays a typed kernarg-derived global pointer so the
    // loads lower to global_load_b128 (LOADcnt), not flat_load (DScnt too).
    int zoff = 0;

    for (int t = 0; t < TT; ++t) {
        __syncthreads();   // x_lds(t) and h_lds(t) ready

        asm volatile("" : "+s"(zoff));          // opaque 0: defeat LICM/hoist
        const bf16_t* WiT = WiT0 + zoff;
        const bf16_t* WhT = WhT0 + zoff;

        v8f accr = {0, 0, 0, 0, 0, 0, 0, 0};  // gi_r + gh_r
        v8f accz = {0, 0, 0, 0, 0, 0, 0, 0};  // gi_z + gh_z
        v8f accg = {0, 0, 0, 0, 0, 0, 0, 0};  // gi_n
        v8f acch = {0, 0, 0, 0, 0, 0, 0, 0};  // gh_n

        // input projection: x[16xDP] @ WiT -> 3 gate columns
        #pragma unroll
        for (int kk = 0; kk < DP / 32; ++kk) {
            v16bf a = loadA(x_lds, DP, kk);
            accr = __builtin_amdgcn_wmma_f32_16x16x32_bf16(
                false, a, false, loadB(WiT, DP, hcol,          kk), (short)0, accr, false, false);
            accz = __builtin_amdgcn_wmma_f32_16x16x32_bf16(
                false, a, false, loadB(WiT, DP, HH + hcol,     kk), (short)0, accz, false, false);
            accg = __builtin_amdgcn_wmma_f32_16x16x32_bf16(
                false, a, false, loadB(WiT, DP, 2 * HH + hcol, kk), (short)0, accg, false, false);
        }
        // hidden projection: h[16xHH] @ WhT -> 3 gate columns
        #pragma unroll
        for (int kk = 0; kk < HH / 32; ++kk) {
            v16bf a = loadA(h_lds, HH, kk);
            accr = __builtin_amdgcn_wmma_f32_16x16x32_bf16(
                false, a, false, loadB(WhT, HH, hcol,          kk), (short)0, accr, false, false);
            accz = __builtin_amdgcn_wmma_f32_16x16x32_bf16(
                false, a, false, loadB(WhT, HH, HH + hcol,     kk), (short)0, accz, false, false);
            acch = __builtin_amdgcn_wmma_f32_16x16x32_bf16(
                false, a, false, loadB(WhT, HH, 2 * HH + hcol, kk), (short)0, acch, false, false);
        }

        // elementwise gates; C layout: VGPR v -> M = v + 8*(lane>>4), N = lane%16
        float hnext[8];
        #pragma unroll
        for (int v = 0; v < 8; ++v) {
            int row = v + (mhi << 3);
            float r = 1.0f / (1.0f + expf(-(accr[v] + bir)));
            float z = 1.0f / (1.0f + expf(-(accz[v] + biz)));
            float n = tanhf(accg[v] + bin + r * (acch[v] + bhnv));
            float hold = h32[row * HH + hcol];
            float nh = (1.0f - z) * n + z * hold;
            int L = L_lds[row];
            hnext[v] = (t >= L) ? hold : nh;   // masked carry
            int ts = dir ? ((L + TT - 1 - t) % TT) : t;
            // reference records the UNMASKED candidate new_h as output;
            // output is write-once streaming -> non-temporal store.
            __builtin_nontemporal_store(
                nh, &out[((size_t)(b0 + row) * TT + ts) * (2 * HH) + dir * HH + hcol]);
        }

        __syncthreads();   // all waves done reading h_lds / x_lds

        #pragma unroll
        for (int v = 0; v < 8; ++v) {
            int row = v + (mhi << 3);
            h32[row * HH + hcol]   = hnext[v];
            h_lds[row * HH + hcol] = (bf16_t)hnext[v];
        }
        if (t + 1 < TT) load_x(t + 1);
    }
}

// ---------------------------------------------------------------------------
extern "C" void kernel_launch(void* const* d_in, const int* in_sizes, int n_in,
                              void* d_out, int out_size, void* d_ws, size_t ws_size,
                              hipStream_t stream) {
    const float* spike = (const float*)d_in[0];
    const float* extin = (const float*)d_in[1];
    const int*   lens  = (const int*)  d_in[2];
    const float* Wi_f  = (const float*)d_in[3];
    const float* Wh_f  = (const float*)d_in[4];
    const float* bi_f  = (const float*)d_in[5];
    const float* bhn_f = (const float*)d_in[6];
    const float* Wi_b  = (const float*)d_in[7];
    const float* Wh_b  = (const float*)d_in[8];
    const float* bi_b  = (const float*)d_in[9];
    const float* bhn_b = (const float*)d_in[10];
    bf16_t* wsb = (bf16_t*)d_ws;   // needs 2*(G3*DP + G3*HH)*2 B ~= 1.25 MB
    float*  out = (float*)d_out;

    gru_prep_weights<<<512, 256, 0, stream>>>(Wi_f, Wh_f, Wi_b, Wh_b, wsb);

    // 16 batch tiles x 2 directions = 32 persistent workgroups
    gru_bidir_kernel<<<BB / BT * 2, NTHREADS, 0, stream>>>(
        spike, extin, lens, bi_f, bhn_f, bi_b, bhn_b, wsb, out);
}
